// TriLinearMatrixAttention_14817637171307
// MI455X (gfx1250) — matrix operand order repaired and checked by reading, not compiled, not run on hardware
//
#include <hip/hip_runtime.h>
#include <hip/hip_bf16.h>
#include <stdint.h>

#define FEAT 768
#define SEQ  2048
#define NB   8

#define BM   128
#define BN   128
#define KT   32
#define LDSS 40   // LDS row stride in halves: 32 data + 4-DWORD TDM pad = 80B (16B multiple)

typedef __attribute__((ext_vector_type(16))) __bf16 v16bf;
typedef __attribute__((ext_vector_type(8)))  float  v8f;
typedef __attribute__((ext_vector_type(4)))  unsigned int v4u;
typedef __attribute__((ext_vector_type(8)))  int v8i;
typedef __attribute__((ext_vector_type(4)))  int v4i;

#if defined(__has_builtin)
#if __has_builtin(__builtin_amdgcn_tensor_load_to_lds) && __has_builtin(__builtin_amdgcn_s_wait_tensorcnt)
#define USE_TDM 1
#endif
#endif

union FragBF {
    v16bf v;
    uint4 q[2];
};

__device__ __forceinline__ unsigned short f32_to_bf16_rne(float x) {
    union { float f; uint32_t u; } c; c.f = x;
    uint32_t u = c.u;
    uint32_t r = u + 0x7fffu + ((u >> 16) & 1u);
    return (unsigned short)(r >> 16);
}

__device__ __forceinline__ unsigned lds_off(const void* p) {
    // Generic LDS pointer: low 32 bits are the LDS byte address (aperture in high bits).
    return (unsigned)(uintptr_t)p;
}

#ifdef USE_TDM
// One TDM descriptor moves a 128x32 bf16 tile (row stride FEAT elements in
// global) into LDS, padding each 64B row with 16B -> 80B LDS row stride.
__device__ __forceinline__ void tdm_load_tile(const unsigned short* gsrc, unsigned lds_addr) {
    unsigned long long ga = (unsigned long long)(uintptr_t)gsrc;
    v4u g0;
    g0[0] = 1u;                                            // count=1 (valid user desc)
    g0[1] = lds_addr;                                      // lds_addr [63:32]
    g0[2] = (unsigned)(ga & 0xffffffffu);                  // global_addr lo
    g0[3] = (unsigned)((ga >> 32) & 0x01ffffffu)           // global_addr [56:32]
          | (2u << 30);                                    // type=2 (image)
    v8i g1;
    g1[0] = (int)((1u << 16)                               // data_size = 2 bytes
          | (1u << 20)                                     // pad_enable
          | (3u << 22)                                     // pad_interval: 16 DWORDs (64B)
          | (3u << 25));                                   // pad_amount: 4 DWORDs (16B)
    g1[1] = (int)(((unsigned)FEAT & 0xffffu) << 16);       // tensor_dim0 lo16 @bits63:48
    g1[2] = (int)((((unsigned)FEAT >> 16) & 0xffffu)       // tensor_dim0 hi16
          | (((unsigned)(NB * SEQ)) & 0xffffu) << 16);     // tensor_dim1 lo16
    g1[3] = (int)(((((unsigned)(NB * SEQ)) >> 16) & 0xffffu)
          | ((unsigned)KT << 16));                         // tile_dim0 = 32
    g1[4] = (int)(unsigned)BM;                             // tile_dim1 = 128, tile_dim2 = 0
    g1[5] = (int)(unsigned)FEAT;                           // tensor_dim0_stride lo32
    g1[6] = 0;                                             // stride hi16 / dim1_stride lo16
    g1[7] = 0;
    v4i z4 = {0, 0, 0, 0};
    v8i z8 = {0, 0, 0, 0, 0, 0, 0, 0};
    // amdgpu-toolchain (clang-23) 6-arg form: (v4u, v8i, v4i, v4i, v8i, i32 cpol)
    __builtin_amdgcn_tensor_load_to_lds(g0, g1, z4, z4, z8, 0);
}
#endif

// One block per row: dot(row, w_dot) and hi/lo bf16 split of (row * w_scale).
__global__ __launch_bounds__(256)
void prep_kernel(const float* __restrict__ src,
                 const float* __restrict__ w_dot,
                 const float* __restrict__ w_scale,
                 float* __restrict__ vec_out,
                 unsigned short* __restrict__ hi_out,
                 unsigned short* __restrict__ lo_out) {
    __shared__ float red[256];
    const int row = blockIdx.x;
    const int tid = threadIdx.x;
    const float* r = src + (size_t)row * FEAT;

    float sum = 0.f;
    for (int f = tid; f < FEAT; f += 256) {
        float x = r[f];
        sum += x * w_dot[f];
        float y = w_scale ? x * w_scale[f] : x;
        union { float f; uint32_t u; } c; c.f = y;
        unsigned short hi = (unsigned short)(c.u >> 16);
        union { uint32_t u; float f; } h; h.u = (uint32_t)hi << 16;
        float lo = y - h.f;
        hi_out[(size_t)row * FEAT + f] = hi;
        lo_out[(size_t)row * FEAT + f] = f32_to_bf16_rne(lo);
    }
    red[tid] = sum;
    __syncthreads();
    for (int s = 128; s > 0; s >>= 1) {
        if (tid < s) red[tid] += red[tid + s];
        __syncthreads();
    }
    if (tid == 0) vec_out[row] = red[0];
}

__global__ __launch_bounds__(256)
void gemm_kernel(const unsigned short* __restrict__ Ah,
                 const unsigned short* __restrict__ Al,
                 const unsigned short* __restrict__ Bh,
                 const unsigned short* __restrict__ Bl,
                 const float* __restrict__ Qv,
                 const float* __restrict__ Dv,
                 float* __restrict__ out) {
    // Double-buffered padded tiles: 2 bufs x 4 tiles x 128 x 40 halves = 80KB.
    __shared__ __align__(16) unsigned short sAh[2][BM][LDSS];
    __shared__ __align__(16) unsigned short sAl[2][BM][LDSS];
    __shared__ __align__(16) unsigned short sBh[2][BN][LDSS];
    __shared__ __align__(16) unsigned short sBl[2][BN][LDSS];

    const int b    = blockIdx.z;
    const int m0   = blockIdx.x * BM;
    const int n0   = blockIdx.y * BN;
    const int tid  = threadIdx.x;
    const int lane = tid & 31;
    const int wave = tid >> 5;
    const int wm   = wave & 3;    // 4 row-groups of 32 rows
    const int wn   = wave >> 2;   // 2 col-groups of 64 cols
    const int hl   = lane >> 4;   // wave32 half
    const int l16  = lane & 15;

    v8f acc[2][4];
    #pragma unroll
    for (int i = 0; i < 2; i++)
        #pragma unroll
        for (int j = 0; j < 4; j++) {
            v8f z = {0.f, 0.f, 0.f, 0.f, 0.f, 0.f, 0.f, 0.f};
            acc[i][j] = z;
        }

    const size_t baseA = ((size_t)b * SEQ + m0) * FEAT;
    const size_t baseB = ((size_t)b * SEQ + n0) * FEAT;

    // --- tile staging -------------------------------------------------------
#ifdef USE_TDM
    #define STAGE(bufi, kk)                                                    \
        if (tid < 32) {                                                        \
            tdm_load_tile(Ah + baseA + (kk), lds_off(&sAh[bufi][0][0]));       \
            tdm_load_tile(Al + baseA + (kk), lds_off(&sAl[bufi][0][0]));       \
            tdm_load_tile(Bh + baseB + (kk), lds_off(&sBh[bufi][0][0]));       \
            tdm_load_tile(Bl + baseB + (kk), lds_off(&sBl[bufi][0][0]));       \
        }
    #define STAGE_WAIT() if (tid < 32) { __builtin_amdgcn_s_wait_tensorcnt(0); }
#else
    #define STAGE(bufi, kk)                                                    \
        _Pragma("unroll")                                                      \
        for (int seg = 0; seg < 2; seg++) {                                    \
            int slot = tid + seg * 256;                                        \
            int row  = slot >> 2;                                              \
            int c16  = (slot & 3) * 8;                                         \
            size_t gA = baseA + (size_t)row * FEAT + (kk) + c16;               \
            size_t gB = baseB + (size_t)row * FEAT + (kk) + c16;               \
            *(uint4*)&sAh[bufi][row][c16] = *(const uint4*)(Ah + gA);          \
            *(uint4*)&sAl[bufi][row][c16] = *(const uint4*)(Al + gA);          \
            *(uint4*)&sBh[bufi][row][c16] = *(const uint4*)(Bh + gB);          \
            *(uint4*)&sBl[bufi][row][c16] = *(const uint4*)(Bl + gB);          \
        }
    #define STAGE_WAIT()
#endif

    int buf = 0;
    STAGE(0, 0)
    STAGE_WAIT()
    __syncthreads();

    for (int kt = 0; kt < FEAT; kt += KT) {
        const int nbuf = buf ^ 1;
        if (kt + KT < FEAT) {
            STAGE(nbuf, kt + KT)   // prefetch next tile while computing current
        }

        // A fragments: row = lane%16; lanes0-15 K{0..15}, lanes16-31 K{16..31}
        FragBF aH[2], aL[2], bH[4], bL[4];
        #pragma unroll
        for (int mi = 0; mi < 2; mi++) {
            int r  = wm * 32 + mi * 16 + l16;
            int k0 = hl * 16;
            aH[mi].q[0] = *(const uint4*)&sAh[buf][r][k0];
            aH[mi].q[1] = *(const uint4*)&sAh[buf][r][k0 + 8];
            aL[mi].q[0] = *(const uint4*)&sAl[buf][r][k0];
            aL[mi].q[1] = *(const uint4*)&sAl[buf][r][k0 + 8];
        }
        // B fragments: column = lane%16 (row of m2), contiguous 16 K per lane
        #pragma unroll
        for (int ni = 0; ni < 4; ni++) {
            int r  = wn * 64 + ni * 16 + l16;
            int k0 = hl * 16;
            bH[ni].q[0] = *(const uint4*)&sBh[buf][r][k0];
            bH[ni].q[1] = *(const uint4*)&sBh[buf][r][k0 + 8];
            bL[ni].q[0] = *(const uint4*)&sBl[buf][r][k0];
            bL[ni].q[1] = *(const uint4*)&sBl[buf][r][k0 + 8];
        }

        #pragma unroll
        for (int mi = 0; mi < 2; mi++)
            #pragma unroll
            for (int ni = 0; ni < 4; ni++) {
                acc[mi][ni] = __builtin_amdgcn_wmma_f32_16x16x32_bf16(
                    false, aH[mi].v, false, bH[ni].v, (short)0, acc[mi][ni], false, false);
                acc[mi][ni] = __builtin_amdgcn_wmma_f32_16x16x32_bf16(
                    false, aH[mi].v, false, bL[ni].v, (short)0, acc[mi][ni], false, false);
                acc[mi][ni] = __builtin_amdgcn_wmma_f32_16x16x32_bf16(
                    false, aL[mi].v, false, bH[ni].v, (short)0, acc[mi][ni], false, false);
            }

        STAGE_WAIT()           // wave0: TDM writes for nbuf complete
        __syncthreads();       // everyone: nbuf visible, buf reads done
        buf = nbuf;
    }

    // Epilogue: C/D layout -> row = r + 8*(lane/16), col = lane%16 within tile.
    const float* Qb = Qv + (size_t)b * SEQ;
    const float* Db = Dv + (size_t)b * SEQ;
    float* outb     = out + (size_t)b * SEQ * SEQ;
    #pragma unroll
    for (int mi = 0; mi < 2; mi++) {
        #pragma unroll
        for (int ni = 0; ni < 4; ni++) {
            int colb = n0 + wn * 64 + ni * 16 + l16;
            float dv = Db[colb];
            #pragma unroll
            for (int rr = 0; rr < 8; rr++) {
                int rowb = m0 + wm * 32 + mi * 16 + hl * 8 + rr;
                outb[(size_t)rowb * SEQ + colb] = acc[mi][ni][rr] + Qb[rowb] + dv;
            }
        }
    }
}

extern "C" void kernel_launch(void* const* d_in, const int* in_sizes, int n_in,
                              void* d_out, int out_size, void* d_ws, size_t ws_size,
                              hipStream_t stream) {
    const float* m1 = (const float*)d_in[0];   // [8,2048,768]
    const float* m2 = (const float*)d_in[1];   // [8,2048,768]
    const float* wq = (const float*)d_in[2];   // [768]
    const float* wd = (const float*)d_in[3];   // [768]
    const float* wc = (const float*)d_in[4];   // [768]
    float* out = (float*)d_out;                // [8,2048,2048]

    const size_t rows       = (size_t)NB * SEQ;
    const size_t mat_halves = rows * FEAT;

    char* ws = (char*)d_ws;
    unsigned short* Ah = (unsigned short*)ws; ws += mat_halves * sizeof(unsigned short);
    unsigned short* Al = (unsigned short*)ws; ws += mat_halves * sizeof(unsigned short);
    unsigned short* Bh = (unsigned short*)ws; ws += mat_halves * sizeof(unsigned short);
    unsigned short* Bl = (unsigned short*)ws; ws += mat_halves * sizeof(unsigned short);
    float* Qv = (float*)ws; ws += rows * sizeof(float);
    float* Dv = (float*)ws;

    prep_kernel<<<dim3((unsigned)rows), 256, 0, stream>>>(m1, wq, wc,      Qv, Ah, Al);
    prep_kernel<<<dim3((unsigned)rows), 256, 0, stream>>>(m2, wd, nullptr, Dv, Bh, Bl);

    dim3 grid(SEQ / BM, SEQ / BN, NB);
    gemm_kernel<<<grid, 256, 0, stream>>>(Ah, Al, Bh, Bl, Qv, Dv, out);
}
